// AGCB_NoGCA_11613591568419
// MI455X (gfx1250) — matrix operand order repaired and checked by reading, not compiled, not run on hardware
//
#include <hip/hip_runtime.h>

typedef __bf16 bf16;
typedef __attribute__((ext_vector_type(16))) __bf16        v16bf;
typedef __attribute__((ext_vector_type(8)))  float         v8f;
typedef __attribute__((ext_vector_type(4)))  unsigned int  u32x4;

union FragB16 {
  v16bf v;
  u32x4 q[2];
  unsigned short u[16];
};

__device__ __forceinline__ v8f wmma_bf16(v16bf a, v16bf b, v8f c) {
  // D(f32 16x16) = A(bf16 16x32) * B(bf16 32x16) + C
  return __builtin_amdgcn_wmma_f32_16x16x32_bf16(false, a, false, b, (short)0, c,
                                                 false, false);
}

// ---- Fragment loaders (contiguous K: k = 16*(lane/16) + s) -------------------------
// A fragment: 16(M) x 32(K), storage row-major [row][K-contiguous]
// lane<16: M=lane, K=0..15; lane>=16: M=lane-16, K=16..31
__device__ __forceinline__ v16bf load_frag_a(const bf16* base, int row0,
                                             int rowStride, int k0, int lane) {
  const bf16* row = base + (size_t)(row0 + (lane & 15)) * rowStride + k0 +
                    ((lane >> 4) & 1) * 16;
  FragB16 f;
  f.q[0] = *(const u32x4*)(row);        // K off .. off+7
  f.q[1] = *(const u32x4*)(row + 8);    // K off+8 .. off+15
  return f.v;
}

// B fragment: 32(K) x 16(N), storage row-major [N-row][K-contiguous]
// lane<16: N=lane, K=0..15 ; lane>=16: N=lane-16, K=16..31
__device__ __forceinline__ v16bf load_frag_bT(const bf16* base, int n0,
                                              int rowStride, int k0, int lane) {
  const bf16* row = base + (size_t)(n0 + (lane & 15)) * rowStride + k0 +
                    ((lane >> 4) & 1) * 16;
  FragB16 f;
  f.q[0] = *(const u32x4*)(row);
  f.q[1] = *(const u32x4*)(row + 8);
  return f.v;
}

// -------------------------------------------------------------------------
static constexpr int B_ = 16, C_ = 256, H_ = 64, W_ = 64;
static constexpr int N_ = 64;        // s*s*B blocks
static constexpr int L_ = 1024;      // 32*32 per block
static constexpr int DPAD = 32;      // d=8 padded to one WMMA K step
static constexpr int KCONV = 9 * C_; // 2304
static constexpr int CP = 264;       // conv LDS channel pitch (16B-aligned rows)

// K0: weight prep: Wv->bf16 ; convW [o][c][3][3] -> bf16 [o][tap*256 + c]
__global__ __launch_bounds__(256) void prep_weights(const float* Wv, const float* convW,
                                                    bf16* Wv_bf, bf16* convW_bf) {
  int gid = blockIdx.x * blockDim.x + threadIdx.x;
  if (gid < C_ * C_) Wv_bf[gid] = (bf16)Wv[gid];
  if (gid < C_ * KCONV) {
    int o = gid / KCONV, k = gid % KCONV;
    int tap = k / C_, c = k % C_;
    convW_bf[gid] = (bf16)convW[((size_t)o * C_ + c) * 9 + tap];
  }
}

// K1: per (n,l): q,k (d=8, zero-padded to 32) in bf16; xb^T in bf16.
__global__ __launch_bounds__(256) void qk_xb_kernel(const float* x, const float* Wq,
                                                    const float* bq, const float* Wk,
                                                    const float* bk, bf16* xbT,
                                                    bf16* q_bf, bf16* kT_bf) {
  __shared__ float sWq[8 * C_], sWk[8 * C_];
  for (int t = threadIdx.x; t < 8 * C_; t += blockDim.x) {
    sWq[t] = Wq[t];
    sWk[t] = Wk[t];
  }
  __syncthreads();
  int gid = blockIdx.x * blockDim.x + threadIdx.x;  // 65536 threads
  int n = gid >> 10, l = gid & 1023;
  int b = n & 15, blk = n >> 4, sh = blk >> 1, sw = blk & 1;
  int h = sh * 32 + (l >> 5), w = sw * 32 + (l & 31);
  const float* xp = x + ((size_t)b * C_ * H_ + h) * W_ + w;  // +c*4096
  bf16* xrow = xbT + ((size_t)n * L_ + l) * C_;
  float qa[8] = {}, ka[8] = {};
  for (int c = 0; c < C_; ++c) {
    float xv = xp[(size_t)c * (H_ * W_)];
    xrow[c] = (bf16)xv;
#pragma unroll
    for (int d = 0; d < 8; ++d) {
      qa[d] += sWq[d * C_ + c] * xv;
      ka[d] += sWk[d * C_ + c] * xv;
    }
  }
  bf16* qr = q_bf + ((size_t)n * L_ + l) * DPAD;
  bf16* kr = kT_bf + ((size_t)n * L_ + l) * DPAD;
#pragma unroll
  for (int d = 0; d < 8; ++d) {
    qr[d] = (bf16)(qa[d] + bq[d]);
    kr[d] = (bf16)(ka[d] + bk[d]);
  }
#pragma unroll
  for (int d = 8; d < DPAD; ++d) {
    qr[d] = (bf16)0.0f;
    kr[d] = (bf16)0.0f;
  }
}

// K2: v = Wv * xb + bv (bf16 WMMA, K=256). grid(64, 8), block 256.
// B tile (128 l x 256 c, 64KB) staged once in LDS and shared by all 8 waves.
__global__ __launch_bounds__(256) void v_gemm_kernel(const bf16* Wv_bf, const bf16* xbT,
                                                     const float* bv, bf16* v_bf) {
  extern __shared__ bf16 sxb[];  // [128][256]
  int n = blockIdx.x, lt = blockIdx.y;
  int lane = threadIdx.x & 31, wv = threadIdx.x >> 5;
  int lbase = lt * 128;
  const bf16* xb_n = xbT + (size_t)n * L_ * C_;

  {  // coalesced b128 stage: 8192 x 16B copied; the 128*256 bf16 tile is 4096 x 16B
    const u32x4* src = (const u32x4*)(xb_n + (size_t)lbase * C_);
    u32x4* dst = (u32x4*)sxb;
    for (int i = threadIdx.x; i < 8192; i += 256) dst[i] = src[i];
  }
  __syncthreads();

  int cbase = wv * 32;
  v8f acc[2][8] = {};
  for (int k0 = 0; k0 < C_; k0 += 32) {
    __builtin_prefetch(Wv_bf + (size_t)(cbase + (lane & 15)) * C_ + k0 + 32, 0, 1);
    v16bf a0 = load_frag_a(Wv_bf, cbase, C_, k0, lane);
    v16bf a1 = load_frag_a(Wv_bf, cbase + 16, C_, k0, lane);
#pragma unroll
    for (int t = 0; t < 8; ++t) {
      v16bf bfr = load_frag_bT(sxb, t * 16, C_, k0, lane);
      acc[0][t] = wmma_bf16(a0, bfr, acc[0][t]);
      acc[1][t] = wmma_bf16(a1, bfr, acc[1][t]);
    }
  }
#pragma unroll
  for (int mt = 0; mt < 2; ++mt)
#pragma unroll
    for (int t = 0; t < 8; ++t)
#pragma unroll
      for (int r = 0; r < 8; ++r) {
        int c = cbase + mt * 16 + r + ((lane >> 4) & 1) * 8;
        int l = lbase + t * 16 + (lane & 15);
        v_bf[((size_t)n * C_ + c) * L_ + l] = (bf16)(acc[mt][t][r] + bv[c]);
      }
}

// K3: fused attention. grid(64, 16) -> (n, 64-row tile). block 256 (8 waves).
// dyn LDS: Ebf[64][1024] bf16 = 128KB. Softmax normalizes in place so the
// phase-3 B fragments are raw 32B LDS vector loads shared by all waves.
__global__ __launch_bounds__(256) void attn_kernel(const bf16* q_bf, const bf16* kT_bf,
                                                   const bf16* v_bf, const float* x,
                                                   const float* gamma_nl, bf16* ctx_bf) {
  extern __shared__ bf16 Ebf[];  // [64][1024]
  int n = blockIdx.x, it = blockIdx.y;
  int lane = threadIdx.x & 31, wv = threadIdx.x >> 5;
  int i0 = it * 64;
  const bf16* qn = q_bf + (size_t)n * L_ * DPAD;
  const bf16* kn = kT_bf + (size_t)n * L_ * DPAD;

  // Phase 1: energy E[i][m] = sum_d q[d,i]k[d,m]; wave wv owns m-range of 128
#pragma unroll
  for (int ii = 0; ii < 4; ++ii) {
    v16bf a = load_frag_a(qn, i0 + ii * 16, DPAD, 0, lane);
#pragma unroll
    for (int mm = 0; mm < 8; ++mm) {
      int m0 = wv * 128 + mm * 16;
      v16bf bfr = load_frag_bT(kn, m0, DPAD, 0, lane);
      v8f acc = {};
      acc = wmma_bf16(a, bfr, acc);
#pragma unroll
      for (int r = 0; r < 8; ++r)
        Ebf[(size_t)(ii * 16 + r + ((lane >> 4) & 1) * 8) * L_ + m0 + (lane & 15)] =
            (bf16)acc[r];
    }
  }
  __syncthreads();

  // Phase 2: row softmax, normalized in place (wave32 butterfly reductions).
  for (int rr = 0; rr < 8; ++rr) {
    bf16* row = Ebf + (size_t)(wv * 8 + rr) * L_;
    float ev[32];
    float mx = -3.4e38f;
#pragma unroll
    for (int jj = 0; jj < 32; ++jj) {
      ev[jj] = (float)row[jj * 32 + lane];
      mx = fmaxf(mx, ev[jj]);
    }
#pragma unroll
    for (int o = 16; o; o >>= 1) mx = fmaxf(mx, __shfl_xor(mx, o, 32));
    float s = 0.f;
#pragma unroll
    for (int jj = 0; jj < 32; ++jj) {
      ev[jj] = __expf(ev[jj] - mx);
      s += ev[jj];
    }
#pragma unroll
    for (int o = 16; o; o >>= 1) s += __shfl_xor(s, o, 32);
    float inv = 1.0f / s;
#pragma unroll
    for (int jj = 0; jj < 32; ++jj) row[jj * 32 + lane] = (bf16)(ev[jj] * inv);
  }
  __syncthreads();

  // Phase 3: out[c,i] = sum_j v[c,j] * P[i,j]; wave wv owns 32 c-rows.
  const bf16* vn = v_bf + (size_t)n * C_ * L_;
  v8f acc[2][4] = {};
  for (int j0 = 0; j0 < L_; j0 += 32) {
    __builtin_prefetch(vn + (size_t)(wv * 32 + (lane & 15)) * L_ + j0 + 32, 0, 1);
    v16bf a0 = load_frag_a(vn, wv * 32, L_, j0, lane);
    v16bf a1 = load_frag_a(vn, wv * 32 + 16, L_, j0, lane);
#pragma unroll
    for (int t = 0; t < 4; ++t) {
      v16bf bfr = load_frag_bT(Ebf, t * 16, L_, j0, lane);  // N=i, K=j
      acc[0][t] = wmma_bf16(a0, bfr, acc[0][t]);
      acc[1][t] = wmma_bf16(a1, bfr, acc[1][t]);
    }
  }
  // Epilogue: ctx = gamma_nl*out + xb (bf16 for conv input)
  float g = gamma_nl[0];
  int b = n & 15, blk = n >> 4, sh = blk >> 1, sw = blk & 1;
#pragma unroll
  for (int mt = 0; mt < 2; ++mt)
#pragma unroll
    for (int t = 0; t < 4; ++t)
#pragma unroll
      for (int r = 0; r < 8; ++r) {
        int c = wv * 32 + mt * 16 + r + ((lane >> 4) & 1) * 8;
        int i = i0 + t * 16 + (lane & 15);
        int h = sh * 32 + (i >> 5), w = sw * 32 + (i & 31);
        size_t off = (((size_t)b * C_ + c) * H_ + h) * W_ + w;
        ctx_bf[off] = (bf16)(g * acc[mt][t][r] + x[off]);
      }
}

// K4: implicit-GEMM 3x3 conv (K=2304) + BN + gamma-residual + ReLU.
// grid(1024) -> (b*64+h). block 256; wave owns 32 o x 64 w.
// Activation slab [3 rows][66 cols][256 ch] staged zero-padded in LDS,
// channel-contiguous so B fragments are two 16B ds loads.
__global__ __launch_bounds__(256) void conv_bn_kernel(const bf16* convW_bf,
                                                      const bf16* ctx_bf,
                                                      const float* convb, const float* bn_w,
                                                      const float* bn_b, const float* bn_mean,
                                                      const float* bn_var, const float* gamma,
                                                      const float* x, float* out) {
  extern __shared__ bf16 sctx[];  // [3*66][CP]
  int bh = blockIdx.x;
  int b = bh >> 6, h = bh & 63;
  int lane = threadIdx.x & 31, wv = threadIdx.x >> 5;
  const bf16* ctx_b = ctx_bf + (size_t)b * C_ * H_ * W_;

  // Stage: r = dh*256 + c ; lanes sweep columns (coalesced global reads).
  for (int r = wv; r < 3 * C_; r += 8) {
    int dh = r >> 8, c = r & 255;
    int hh = h + dh - 1;
    bool hok = (unsigned)hh < (unsigned)H_;
    const bf16* src = ctx_b + ((size_t)c * H_ + (hok ? hh : 0)) * W_;
    for (int col = lane; col < 66; col += 32) {
      int ww = col - 1;
      bf16 vv = (hok && (unsigned)ww < (unsigned)W_) ? src[ww] : (bf16)0.0f;
      sctx[(size_t)(dh * 66 + col) * CP + c] = vv;
    }
  }
  __syncthreads();

  int obase = wv * 32;
  v8f acc[2][4] = {};
  for (int kk = 0; kk < 72; ++kk) {
    int tap = kk >> 3;
    int dh = tap / 3, dw = tap % 3;
    int c0 = (kk & 7) * 32;
    int ka = tap * C_ + c0;
    __builtin_prefetch(convW_bf + (size_t)(obase + (lane & 15)) * KCONV + ka + 32, 0, 1);
    v16bf a0 = load_frag_a(convW_bf, obase, KCONV, ka, lane);
    v16bf a1 = load_frag_a(convW_bf, obase + 16, KCONV, ka, lane);
#pragma unroll
    for (int t = 0; t < 4; ++t) {
      // B frag: N = pixel w (col idx = w+dw), K = channel c0..c0+31
      const bf16* rowp = sctx + (size_t)(dh * 66 + dw + t * 16 + (lane & 15)) * CP + c0 +
                         ((lane >> 4) & 1) * 16;
      FragB16 f;
      f.q[0] = *(const u32x4*)rowp;
      f.q[1] = *(const u32x4*)(rowp + 8);
      acc[0][t] = wmma_bf16(a0, f.v, acc[0][t]);
      acc[1][t] = wmma_bf16(a1, f.v, acc[1][t]);
    }
  }
  float g = gamma[0];
#pragma unroll
  for (int mt = 0; mt < 2; ++mt)
#pragma unroll
    for (int t = 0; t < 4; ++t)
#pragma unroll
      for (int r = 0; r < 8; ++r) {
        int o = obase + mt * 16 + r + ((lane >> 4) & 1) * 8;
        int w = t * 16 + (lane & 15);
        float y = acc[mt][t][r] + convb[o];
        float inv = bn_w[o] * rsqrtf(bn_var[o] + 1e-5f);
        y = (y - bn_mean[o]) * inv + bn_b[o];
        size_t off = (((size_t)b * C_ + o) * H_ + h) * W_ + w;
        out[off] = fmaxf(g * y + x[off], 0.0f);
      }
}

// -------------------------------------------------------------------------
extern "C" void kernel_launch(void* const* d_in, const int* in_sizes, int n_in,
                              void* d_out, int out_size, void* d_ws, size_t ws_size,
                              hipStream_t stream) {
  (void)in_sizes; (void)n_in; (void)out_size;
  const float* x        = (const float*)d_in[0];
  const float* Wq       = (const float*)d_in[1];
  const float* bq       = (const float*)d_in[2];
  const float* Wk       = (const float*)d_in[3];
  const float* bk       = (const float*)d_in[4];
  const float* Wv       = (const float*)d_in[5];
  const float* bv       = (const float*)d_in[6];
  const float* gamma_nl = (const float*)d_in[7];
  const float* convW    = (const float*)d_in[8];
  const float* convb    = (const float*)d_in[9];
  const float* bn_w     = (const float*)d_in[10];
  const float* bn_b     = (const float*)d_in[11];
  const float* bn_mean  = (const float*)d_in[12];
  const float* bn_var   = (const float*)d_in[13];
  const float* gamma    = (const float*)d_in[14];
  float* out = (float*)d_out;

  // Workspace layout (all 256B aligned)
  size_t off = 0;
  auto take = [&](size_t bytes) {
    size_t o = off;
    off += (bytes + 255) & ~(size_t)255;
    return o;
  };
  char* ws = (char*)d_ws;
  size_t o_xbT  = take((size_t)N_ * L_ * C_ * 2);    // 33.5 MB
  size_t o_q    = take((size_t)N_ * L_ * DPAD * 2);  // 4 MB
  size_t o_kT   = take((size_t)N_ * L_ * DPAD * 2);  // 4 MB
  size_t o_Wv   = take((size_t)C_ * C_ * 2);
  size_t o_v    = take((size_t)N_ * C_ * L_ * 2);    // 33.5 MB
  size_t o_ctx  = take((size_t)B_ * C_ * H_ * W_ * 2);
  size_t o_cw   = take((size_t)C_ * KCONV * 2);
  if (off > ws_size) return;  // workspace too small: no work (deterministic)

  bf16* xbT_bf   = (bf16*)(ws + o_xbT);
  bf16* q_bf     = (bf16*)(ws + o_q);
  bf16* kT_bf    = (bf16*)(ws + o_kT);
  bf16* Wv_bf    = (bf16*)(ws + o_Wv);
  bf16* v_bf     = (bf16*)(ws + o_v);
  bf16* ctx_bf   = (bf16*)(ws + o_ctx);
  bf16* convW_bf = (bf16*)(ws + o_cw);

  prep_weights<<<(C_ * KCONV + 255) / 256, 256, 0, stream>>>(Wv, convW, Wv_bf, convW_bf);
  qk_xb_kernel<<<(N_ * L_) / 256, 256, 0, stream>>>(x, Wq, bq, Wk, bk, xbT_bf, q_bf, kT_bf);
  v_gemm_kernel<<<dim3(N_, 8), 256, (size_t)128 * C_ * 2, stream>>>(Wv_bf, xbT_bf, bv, v_bf);
  attn_kernel<<<dim3(N_, 16), 256, (size_t)64 * L_ * 2, stream>>>(q_bf, kT_bf, v_bf, x,
                                                                  gamma_nl, ctx_bf);
  conv_bn_kernel<<<dim3(B_ * H_), 256, (size_t)3 * 66 * CP * 2, stream>>>(
      convW_bf, ctx_bf, convb, bn_w, bn_b, bn_mean, bn_var, gamma, x, out);
}
